// SelfAttention1D_64819646431675
// MI455X (gfx1250) — matrix operand order repaired and checked by reading, not compiled, not run on hardware
//
#include <hip/hip_runtime.h>
#include <hip/hip_bf16.h>
#include <math.h>

// Problem constants (match reference)
#define BATCH 4
#define SEQ   4096
#define DIN   256
#define DF    64    // d_feat == d_out

typedef __attribute__((ext_vector_type(16))) __bf16 v16bf;
typedef __attribute__((ext_vector_type(8)))  float  v8f;

// TDM descriptor vector types (probe-verified builtin signatures)
typedef __attribute__((ext_vector_type(4))) unsigned tdm_u4;
typedef __attribute__((ext_vector_type(8))) int      tdm_i8;
typedef __attribute__((ext_vector_type(4))) int      tdm_i4;

__device__ __forceinline__ v8f v8f_zero() {
    v8f z;
#pragma unroll
    for (int i = 0; i < 8; ++i) z[i] = 0.f;
    return z;
}

__device__ __forceinline__ float hmax16(float v) {
#pragma unroll
    for (int m = 8; m >= 1; m >>= 1) v = fmaxf(v, __shfl_xor(v, m, 32));
    return v;  // xor masks < 16 keep the reduction inside each 16-lane half
}
__device__ __forceinline__ float hsum16(float v) {
#pragma unroll
    for (int m = 8; m >= 1; m >>= 1) v += __shfl_xor(v, m, 32);
    return v;
}

// integer power by squaring (reference: scores ** power, power is a runtime int)
__device__ __forceinline__ float ipow(float s, int p) {
    float r = 1.f, b = s;
    int e = p;
    while (e > 0) { if (e & 1) r *= b; b *= b; e >>= 1; }
    return r;
}

// ---------------------------------------------------------------------------
// Kernel 1: Q/K/V projections as a WMMA GEMM.
// x[16384,256] (fp32 -> bf16 on the fly) @ W[256,64] x3 -> bf16 workspace.
// Block = 128 threads = 4 waves; block covers 64 rows; wave covers 16 rows.
// A fragments (8 K-chunks of 32) live in registers for all three projections;
// each W is staged transposed in LDS so the B-operand is contiguous per lane.
// ---------------------------------------------------------------------------
__global__ __launch_bounds__(128) void proj_kernel(
    const float* __restrict__ x,
    const float* __restrict__ wq, const float* __restrict__ wk,
    const float* __restrict__ wv,
    __bf16* __restrict__ qbf, __bf16* __restrict__ kbf, __bf16* __restrict__ vbf)
{
    __shared__ __align__(16) __bf16 wt[DF * DIN];   // W^T tile: [col][k], 32 KB

    const size_t rowbase = (size_t)blockIdx.x * 64;
    const int wid = threadIdx.x >> 5;
    const int lane = threadIdx.x & 31;
    const int hl = lane & 15;
    const int hi = lane >> 4;
    const int row = (int)(rowbase) + wid * 16 + hl;   // A-layout: lane row = lane%16

    // ---- A fragments: 16x32 bf16 per chunk, 8 chunks cover K=256 ----
    v16bf xa[8];
    {
        const float* xp = x + (size_t)row * DIN;
#pragma unroll
        for (int kc = 0; kc < 8; ++kc) {
            const int k0 = 32 * kc + 8 * hi;
#pragma unroll
            for (int t = 0; t < 8; ++t) {
                xa[kc][t]     = (__bf16)xp[k0 + t];
                xa[kc][t + 8] = (__bf16)xp[k0 + 16 + t];
            }
        }
    }

    const float* wsel[3] = { wq, wk, wv };
    __bf16* dsel[3] = { qbf, kbf, vbf };

    for (int p = 0; p < 3; ++p) {
        __syncthreads();   // previous phase done reading wt
        // stage W^T: read w[k*64+col] coalesced, write wt[col*256+k] as bf16
        const float* w = wsel[p];
        for (int idx = threadIdx.x; idx < DF * DIN; idx += 128) {
            const int col = idx & (DF - 1), k = idx >> 6;
            wt[col * DIN + k] = (__bf16)w[(size_t)k * DF + col];
        }
        __syncthreads();

        __bf16* dst = dsel[p];
#pragma unroll
        for (int c = 0; c < 4; ++c) {                 // 4 col-chunks of 16
            v8f acc = v8f_zero();
#pragma unroll
            for (int kc = 0; kc < 8; ++kc) {          // K = 256 in chunks of 32
                // B layout (32x16): lane col N = lane%16; K = 8*hi+t (t<8), 16+8*hi+(t-8) (t>=8)
                const __bf16* bp = &wt[(c * 16 + hl) * DIN + 32 * kc + 8 * hi];
                v16bf b;
#pragma unroll
                for (int t = 0; t < 8; ++t) { b[t] = bp[t]; b[t + 8] = bp[16 + t]; }
                acc = __builtin_amdgcn_wmma_f32_16x16x32_bf16(false, xa[kc], false, b,
                                                              (short)0, acc, false, false);
            }
            // C layout: VGPR j -> row j+8*hi, col hl (within this 16-col chunk)
#pragma unroll
            for (int j = 0; j < 8; ++j)
                dst[(rowbase + (size_t)wid * 16 + j + 8 * hi) * DF + c * 16 + hl] =
                    (__bf16)acc[j];
        }
    }
}

// ---------------------------------------------------------------------------
// Kernel 2: flash attention with (q.k)^power scores.
// Block = 128 threads = 4 waves, 64 query rows (16-row strip per wave).
// K tile staged to LDS by the Tensor Data Mover; V tile staged transposed.
// ---------------------------------------------------------------------------
__global__ __launch_bounds__(128) void attn_kernel(
    const __bf16* __restrict__ qbf, const __bf16* __restrict__ kbf,
    const __bf16* __restrict__ vbf, float* __restrict__ out,
    const int* __restrict__ power_p)
{
    __shared__ __align__(16) __bf16 kt[64 * DF];      // K tile, [key][feat]
    __shared__ __align__(16) __bf16 vt[DF * 64];      // V tile transposed, [feat][key]
    __shared__ __align__(16) __bf16 pt[4 * 16 * 64];  // per-wave P strips

    const int b   = blockIdx.x >> 6;          // SEQ/64 == 64 tiles per batch
    const int qt  = blockIdx.x & 63;
    const int wid = threadIdx.x >> 5;
    const int lane = threadIdx.x & 31;
    const int hl = lane & 15;                 // half-lane
    const int hi = lane >> 4;                 // which 16-lane half

    const int power = *power_p;
    const size_t bbase = (size_t)b * SEQ * DF;
    const int qrow0 = qt * 64 + wid * 16;

    // ---- Q strip (16 x 64) into WMMA A-layout registers (2 chunks of 16x32) ----
    v16bf qa[2];
    {
        const __bf16* qp = qbf + bbase + (size_t)(qrow0 + hl) * DF;
#pragma unroll
        for (int c = 0; c < 2; ++c) {
            const int k0 = 32 * c + 8 * hi;
#pragma unroll
            for (int t = 0; t < 8; ++t) {
                qa[c][t]     = qp[k0 + t];
                qa[c][t + 8] = qp[k0 + 16 + t];
            }
        }
    }

    // Online-softmax state. C-layout: VGPR j of lane (hi,hl) is row j+8*hi, col hl.
    v8f o[4];
#pragma unroll
    for (int c = 0; c < 4; ++c) o[c] = v8f_zero();
    float rmax[8], rsum[8];
#pragma unroll
    for (int j = 0; j < 8; ++j) { rmax[j] = -INFINITY; rsum[j] = 0.f; }

    const unsigned kt_lds = (unsigned)(uintptr_t)&kt[0];  // low 32 bits = LDS offset

    for (int jt = 0; jt < SEQ / 64; ++jt) {
        __syncthreads();  // previous tile fully consumed before restaging

        // ---- stage K tile via Tensor Data Mover (wave 0 issues one DMA) ----
#if defined(__gfx1250__)
        if (wid == 0) {
            const unsigned long long ga =
                (unsigned long long)(uintptr_t)(kbf + bbase + (size_t)jt * 64 * DF);
            tdm_u4 g0;
            g0[0] = 1u;                                  // count=1, user descriptor
            g0[1] = kt_lds;                              // lds_addr
            g0[2] = (unsigned)(ga & 0xFFFFFFFFu);        // global_addr[31:0]
            g0[3] = (unsigned)((ga >> 32) & 0x01FFFFFFu) // global_addr[56:32]
                    | 0x80000000u;                       // type=2 ("image")
            tdm_i8 g1;
            g1[0] = 0x00010000;          // data_size=1 (2B elems)
            g1[1] = (int)(64u << 16);    // tensor_dim0 = 64   (bits 79:48)
            g1[2] = (int)(64u << 16);    // tensor_dim1 = 64   (bits 111:80)
            g1[3] = (int)(64u << 16);    // tile_dim0   = 64   (bits 127:112)
            g1[4] = 64;                  // tile_dim1   = 64   (bits 143:128)
            g1[5] = 64;                  // tensor_dim0_stride = 64 (bits 207:160)
            g1[6] = 0;
            g1[7] = 0;
            tdm_i4 z4; z4[0] = z4[1] = z4[2] = z4[3] = 0;
#if __clang_major__ >= 23
            tdm_i8 z8;
#pragma unroll
            for (int z = 0; z < 8; ++z) z8[z] = 0;
            __builtin_amdgcn_tensor_load_to_lds(g0, g1, z4, z4, z8, 0);
#else
            __builtin_amdgcn_tensor_load_to_lds(g0, g1, z4, z4, 0);
#endif
        }
#endif
        // ---- all threads stage V^T (so PV B-operand is contiguous per lane) ----
        const __bf16* vp = vbf + bbase + (size_t)jt * 64 * DF;
        __builtin_prefetch((const void*)(vp + 64 * DF), 0, 1);  // next V tile
        for (int idx = threadIdx.x; idx < 64 * DF; idx += 128) {
            const int key = idx >> 6, f = idx & 63;
            vt[f * 64 + key] = vp[idx];
        }
#if defined(__gfx1250__)
        if (wid == 0) __builtin_amdgcn_s_wait_tensorcnt(0);
#endif
        __syncthreads();  // kt (TDM) + vt visible to everyone

        // ---- S = Q K^T : 4 column chunks of 16 keys, K-dim 64 = 2 chained WMMAs ----
        v8f s[4];
#pragma unroll
        for (int c = 0; c < 4; ++c) {
            const __bf16* kp = &kt[(c * 16 + hl) * DF + 8 * hi];
            v16bf kb0, kb1;
#pragma unroll
            for (int t = 0; t < 8; ++t) {
                kb0[t] = kp[t]; kb0[t + 8] = kp[16 + t];
                kb1[t] = kp[32 + t]; kb1[t + 8] = kp[48 + t];
            }
            v8f acc = v8f_zero();
            acc = __builtin_amdgcn_wmma_f32_16x16x32_bf16(false, qa[0], false, kb0,
                                                          (short)0, acc, false, false);
            acc = __builtin_amdgcn_wmma_f32_16x16x32_bf16(false, qa[1], false, kb1,
                                                          (short)0, acc, false, false);
            s[c] = acc;
        }

        // ---- powered scores + online softmax update ----
        float tmax[8];
#pragma unroll
        for (int j = 0; j < 8; ++j) tmax[j] = -INFINITY;
#pragma unroll
        for (int c = 0; c < 4; ++c)
#pragma unroll
            for (int j = 0; j < 8; ++j) {
                const float p = ipow(s[c][j], power);
                s[c][j] = p;
                tmax[j] = fmaxf(tmax[j], p);
            }
#pragma unroll
        for (int j = 0; j < 8; ++j) tmax[j] = hmax16(tmax[j]);

        float alpha[8];
#pragma unroll
        for (int j = 0; j < 8; ++j) {
            const float nm = fmaxf(rmax[j], tmax[j]);
            alpha[j] = __expf(rmax[j] - nm);   // exp(-inf)=0 on first tile
            rmax[j] = nm;
        }
        float tsum[8];
#pragma unroll
        for (int j = 0; j < 8; ++j) tsum[j] = 0.f;
#pragma unroll
        for (int c = 0; c < 4; ++c)
#pragma unroll
            for (int j = 0; j < 8; ++j) {
                const float p = __expf(s[c][j] - rmax[j]);
                s[c][j] = p;
                tsum[j] += p;
            }
#pragma unroll
        for (int j = 0; j < 8; ++j) {
            tsum[j] = hsum16(tsum[j]);
            rsum[j] = rsum[j] * alpha[j] + tsum[j];
        }
#pragma unroll
        for (int c = 0; c < 4; ++c)
#pragma unroll
            for (int j = 0; j < 8; ++j) o[c][j] *= alpha[j];

        // ---- C-layout P -> bf16 A-layout via per-wave LDS strip ----
        __bf16* ptw = &pt[wid * 16 * 64];
#pragma unroll
        for (int c = 0; c < 4; ++c)
#pragma unroll
            for (int j = 0; j < 8; ++j)
                ptw[(j + 8 * hi) * 64 + c * 16 + hl] = (__bf16)s[c][j];
        asm volatile("s_wait_dscnt 0" ::: "memory");  // cross-lane via LDS, same wave

        v16bf pa[2];
        {
            const __bf16* pp = ptw + hl * 64;
#pragma unroll
            for (int c = 0; c < 2; ++c) {
                const int k0 = 32 * c + 8 * hi;
#pragma unroll
                for (int t = 0; t < 8; ++t) {
                    pa[c][t]     = pp[k0 + t];
                    pa[c][t + 8] = pp[k0 + 16 + t];
                }
            }
        }

        // ---- O += P V : B-operand from transposed V tile (contiguous per lane) ----
#pragma unroll
        for (int c = 0; c < 4; ++c) {
            const __bf16* vpp = &vt[(c * 16 + hl) * 64 + 8 * hi];
            v16bf vb0, vb1;
#pragma unroll
            for (int t = 0; t < 8; ++t) {
                vb0[t] = vpp[t]; vb0[t + 8] = vpp[16 + t];
                vb1[t] = vpp[32 + t]; vb1[t + 8] = vpp[48 + t];
            }
            o[c] = __builtin_amdgcn_wmma_f32_16x16x32_bf16(false, pa[0], false, vb0,
                                                           (short)0, o[c], false, false);
            o[c] = __builtin_amdgcn_wmma_f32_16x16x32_bf16(false, pa[1], false, vb1,
                                                           (short)0, o[c], false, false);
        }
    }

    // ---- epilogue: normalize and store (C layout scatter) ----
    float* op = out + bbase + (size_t)qrow0 * DF;
#pragma unroll
    for (int j = 0; j < 8; ++j) {
        const float inv = 1.f / rsum[j];
#pragma unroll
        for (int c = 0; c < 4; ++c)
            op[(j + 8 * hi) * DF + c * 16 + hl] = o[c][j] * inv;
    }
}

// ---------------------------------------------------------------------------
// Launcher. Workspace layout: Q | K | V as bf16, 2 MB each (needs 6 MB of ws).
// ---------------------------------------------------------------------------
extern "C" void kernel_launch(void* const* d_in, const int* in_sizes, int n_in,
                              void* d_out, int out_size, void* d_ws, size_t ws_size,
                              hipStream_t stream) {
    (void)in_sizes; (void)n_in; (void)out_size; (void)ws_size;
    const float* x  = (const float*)d_in[0];
    const float* wq = (const float*)d_in[1];
    const float* wk = (const float*)d_in[2];
    const float* wv = (const float*)d_in[3];
    const int*   pw = (const int*)d_in[4];
    float* out = (float*)d_out;

    const size_t n = (size_t)BATCH * SEQ * DF;
    __bf16* qbf = (__bf16*)d_ws;
    __bf16* kbf = qbf + n;
    __bf16* vbf = kbf + n;

    proj_kernel<<<(BATCH * SEQ) / 64, 128, 0, stream>>>(x, wq, wk, wv, qbf, kbf, vbf);
    attn_kernel<<<BATCH * (SEQ / 64), 128, 0, stream>>>(qbf, kbf, vbf, out, pw);
}
